// NewsEncoder_61220463837567
// MI455X (gfx1250) — matrix operand order repaired and checked by reading, not compiled, not run on hardware
//
#include <hip/hip_runtime.h>
#include <hip/hip_bf16.h>
#include <math.h>

typedef float    v2f  __attribute__((ext_vector_type(2)));
typedef float    v8f  __attribute__((ext_vector_type(8)));
typedef _Float16 v16h __attribute__((ext_vector_type(16)));

#define BSN   800        // bs*n
#define TS    128
#define D     768
#define HID   512
#define NCAT  15
#define NNER  9
#define KS1   (D/4)      // 192 K-steps for f32 GEMM1 (cat head)
#define KC1   (D/32)     // 24  K-chunks for f16 GEMM1 (ner head)
#define NC1   (HID/16)   // 32  N-chunks of hid
#define KS2   (HID/4)    // 128 K-steps for f32 GEMM2

// ---- output layout (floats, concatenated in reference return order) ----
#define OUT_EMB   0
#define OUT_CAT   (BSN*D)                    // 614400
#define OUT_NER   (OUT_CAT + BSN*NCAT)       // 626400
#define OUT_MASK  (OUT_NER + BSN*TS*NNER)    // 1548000

// ---- workspace layout (bytes) ----
#define WSB_NW1H  0                          // ner W1 hi (f16 frags)  768 KB
#define WSB_NW1L  786432                     // ner W1 lo (f16 frags)  768 KB
#define WSB_CW1   1572864                    // cat W1 (f32 frags)     1.5 MB
#define WSB_NW2   3145728                    // ner W2 (f32 frags)     32 KB
#define WSB_CW2   3178496                    // cat W2 (f32 frags)     32 KB

static __device__ __forceinline__ v8f wmma_f32(v2f a, v2f b, v8f c) {
  return __builtin_amdgcn_wmma_f32_16x16x4_f32(false, a, false, b, (short)0, c,
                                               false, false);
}
static __device__ __forceinline__ v8f wmma_f16(v16h a, v16h b, v8f c) {
  return __builtin_amdgcn_wmma_f32_16x16x32_f16(false, a, false, b, (short)0, c,
                                                false, false);
}

// Split 16 f32 (two 8-float runs) into f16 hi/lo A-fragments.
static __device__ __forceinline__ void cvt_hilo(const float* __restrict__ p0,
                                                const float* __restrict__ p1,
                                                v16h& hi, v16h& lo) {
  float4 q0 = *(const float4*)(p0);
  float4 q1 = *(const float4*)(p0 + 4);
  float4 q2 = *(const float4*)(p1);
  float4 q3 = *(const float4*)(p1 + 4);
  float f[16] = {q0.x,q0.y,q0.z,q0.w, q1.x,q1.y,q1.z,q1.w,
                 q2.x,q2.y,q2.z,q2.w, q3.x,q3.y,q3.z,q3.w};
  #pragma unroll
  for (int j = 0; j < 16; ++j) {
    _Float16 h = (_Float16)f[j];
    hi[j] = h;
    lo[j] = (_Float16)(f[j] - (float)h);
  }
}

// ---- Pack W1 (D x HID) into f16 hi/lo B-fragment layout for 16x16x32 ----
// frag(kc,nc): 32 lanes x 16 halves; lane holds B[K = kc*32 + (lane>=16?16:0)+j]
// [N = nc*16 + lane%16], j=0..15.  Frag stride 512 halves, lane stride 16.
__global__ void pack_w1_f16_kernel(const float* __restrict__ W1,
                                   _Float16* __restrict__ hi,
                                   _Float16* __restrict__ lo) {
  int t = blockIdx.x * blockDim.x + threadIdx.x;
  if (t >= KC1 * NC1 * 32) return;
  int lane = t & 31;
  int nc   = (t >> 5) & 31;
  int kc   = t >> 10;
  int n     = nc * 16 + (lane & 15);
  int kbase = kc * 32 + ((lane >> 4) << 4);
  size_t base = ((size_t)(kc * 32 + nc)) * 512 + lane * 16;
  #pragma unroll 4
  for (int j = 0; j < 16; ++j) {
    float f = W1[(size_t)(kbase + j) * HID + n];
    _Float16 h = (_Float16)f;
    hi[base + j] = h;
    lo[base + j] = (_Float16)(f - (float)h);
  }
}

// ---- Pack W1 (D x HID) into f32 B-fragment layout for 16x16x4 (cat head) ----
__global__ void pack_w1_f32_kernel(const float* __restrict__ W1,
                                   float* __restrict__ dst) {
  int t = blockIdx.x * blockDim.x + threadIdx.x;
  if (t >= KS1 * NC1 * 32) return;
  int lane = t & 31;
  int nc   = (t >> 5) & 31;
  int ks   = t >> 10;
  int k = ks * 4 + ((lane >> 4) << 1);
  int n = nc * 16 + (lane & 15);
  float2 v;
  v.x = W1[(size_t)k * HID + n];
  v.y = W1[(size_t)(k + 1) * HID + n];
  ((float2*)dst)[((size_t)ks * 32 + nc) * 32 + lane] = v;
}

// ---- Pack W2 (HID x ncols) into f32 B-fragments, zero-padded to 16 cols ----
__global__ void pack_w2_kernel(const float* __restrict__ W2, int ncols,
                               float* __restrict__ dst) {
  int t = blockIdx.x * blockDim.x + threadIdx.x;
  if (t >= KS2 * 32) return;
  int lane = t & 31;
  int ks   = t >> 5;
  int k = ks * 4 + ((lane >> 4) << 1);
  int c = lane & 15;
  float2 v;
  v.x = (c < ncols) ? W2[(size_t)k * ncols + c] : 0.f;
  v.y = (c < ncols) ? W2[(size_t)(k + 1) * ncols + c] : 0.f;
  ((float2*)dst)[(size_t)ks * 32 + lane] = v;
}

// One workgroup per news item: mask_ner + news_emb + fused 2-layer NER MLP.
// Layer 1: split-f16 WMMA (3 terms, fp32-class accuracy); layer 2: f32 WMMA.
__global__ __launch_bounds__(256) void ner_kernel(
    const float* __restrict__ lhs, const int* __restrict__ mask,
    const float* __restrict__ nb1, const float* __restrict__ nb2,
    const _Float16* __restrict__ w1h, const _Float16* __restrict__ w1l,
    const float* __restrict__ w2p, float* __restrict__ out)
{
  __shared__ float hbuf[8][320];   // per-wave 16x16 transpose bounce, pitch 20
  __shared__ float logits[TS][10];
  __shared__ int   mn[TS];
  __shared__ int   validFlag;

  const int item = blockIdx.x;
  const int tid  = threadIdx.x;
  const int wave = tid >> 5;
  const int lane = tid & 31;

  if (tid < TS) mn[tid] = mask[item * TS + tid];
  __syncthreads();
  if (tid == 0) {
    int s = 0;
    for (int i = 0; i < TS; ++i) s += mn[i];
    validFlag = (s > 0);
    mn[0] = 0;                                   // m[:,0] = 0
    for (int i = TS - 1; i >= 0; --i) {          // clear last remaining 1
      if (mn[i]) { mn[i] = 0; break; }
    }
  }
  __syncthreads();
  const int valid = validFlag;

  if (tid < TS) out[OUT_MASK + item * TS + tid] = (float)mn[tid];

  const float* X = lhs + (size_t)item * TS * D;
  for (int i = tid; i < D; i += 256)
    out[OUT_EMB + item * D + i] = valid ? X[i] : 0.f;

  if (!valid) {
    const float nanv = __int_as_float(0x7FC00000);
    for (int i = tid; i < TS * NNER; i += 256)
      out[OUT_NER + (size_t)item * TS * NNER + i] = nanv;
    return;
  }

  const int   col   = lane & 15;          // A-row selector AND C-column
  const int   kadd  = (lane >> 4) << 1;   // 0/2 (f32 frags)
  const int   off8  = (lane >> 4) << 3;   // 0/8 (f16 frags)
  const int   Mbase = wave * 16;
  const int   Mloc0 = (lane < 16) ? 0 : 8;
  const float* Arow = X + (size_t)(Mbase + col) * D;
  const v2f*  w2v   = (const v2f*)w2p;
  float*      hb    = hbuf[wave];

  const v8f zero8 = {0.f, 0.f, 0.f, 0.f, 0.f, 0.f, 0.f, 0.f};
  v8f nacc = zero8;

  for (int ng = 0; ng < 4; ++ng) {        // 4 groups of 8 hid-chunks
    v8f hacc[8];
    #pragma unroll
    for (int i = 0; i < 8; ++i) hacc[i] = zero8;

    for (int kc = 0; kc < KC1; ++kc) {    // K loop over d=768, 32 per chunk
      v16h ah, al;
      cvt_hilo(Arow + kc * 32 + 2 * off8, Arow + kc * 32 + 2 * off8 + 8, ah, al);
      #pragma unroll
      for (int i = 0; i < 8; ++i) {
        size_t fi = (size_t)(kc * 32 + ng * 8 + i) * 512 + lane * 16;
        v16h bh = *(const v16h*)(w1h + fi);
        v16h bl = *(const v16h*)(w1l + fi);
        hacc[i] = wmma_f16(ah, bh, hacc[i]);  // hi*hi
        hacc[i] = wmma_f16(al, bh, hacc[i]);  // lo*hi
        hacc[i] = wmma_f16(ah, bl, hacc[i]);  // hi*lo
      }
    }

    // bias + ReLU, transpose via LDS, feed exact f32 layer-2 accumulator
    for (int i = 0; i < 8; ++i) {
      const int   ncg  = ng * 8 + i;
      const float bias = nb1[ncg * 16 + col];
      #pragma unroll
      for (int r = 0; r < 8; ++r)
        hb[(Mloc0 + r) * 20 + col] = fmaxf(hacc[i][r] + bias, 0.f);
      #pragma unroll
      for (int k2 = 0; k2 < 4; ++k2) {
        v2f a2 = *(const v2f*)(hb + col * 20 + k2 * 4 + kadd);
        v2f b2 = w2v[(size_t)(ncg * 4 + k2) * 32 + lane];
        nacc = wmma_f32(a2, b2, nacc);
      }
    }
  }

  if (col < NNER) {
    #pragma unroll
    for (int r = 0; r < 8; ++r)
      logits[Mbase + Mloc0 + r][col] = nacc[r] + nb2[col];
  }
  __syncthreads();

  if (tid < TS) {  // one thread per token: masked softmax over 9 classes
    float* o = out + OUT_NER + (size_t)item * TS * NNER + tid * NNER;
    if (mn[tid] == 0) {
      const float nanv = __int_as_float(0x7FC00000);
      for (int c = 0; c < NNER; ++c) o[c] = nanv;
    } else {
      float mx = -3.4e38f;
      for (int c = 0; c < NNER; ++c) mx = fmaxf(mx, logits[tid][c]);
      float e[NNER], s = 0.f;
      for (int c = 0; c < NNER; ++c) { e[c] = __expf(logits[tid][c] - mx); s += e[c]; }
      float inv = 1.f / s;
      for (int c = 0; c < NNER; ++c) o[c] = e[c] * inv;
    }
  }
}

// One wave per 16 news items: CAT head on token-0 rows (exact f32 WMMA path).
__global__ __launch_bounds__(32) void cat_kernel(
    const float* __restrict__ lhs, const int* __restrict__ mask,
    const float* __restrict__ cb1, const float* __restrict__ cb2,
    const float* __restrict__ w1p, const float* __restrict__ w2p,
    float* __restrict__ out)
{
  __shared__ float hb[320];
  __shared__ float lg[16][16];
  __shared__ int   vflag[16];

  const int g0    = blockIdx.x * 16;
  const int lane  = threadIdx.x;
  const int col   = lane & 15;
  const int kadd  = (lane >> 4) << 1;
  const int Mloc0 = (lane < 16) ? 0 : 8;

  if (lane < 16) {
    int s = 0;
    const int* mr = mask + (size_t)(g0 + lane) * TS;
    for (int i = 0; i < TS; ++i) s += mr[i];
    vflag[lane] = (s > 0);
  }
  __syncthreads();

  const float* Arow = lhs + (size_t)(g0 + col) * TS * D + kadd;  // token 0 row
  const v2f*  w1v = (const v2f*)w1p;
  const v2f*  w2v = (const v2f*)w2p;
  const v8f zero8 = {0.f, 0.f, 0.f, 0.f, 0.f, 0.f, 0.f, 0.f};
  v8f cacc = zero8;

  for (int nc = 0; nc < NC1; ++nc) {
    v8f h = zero8;
    for (int ks = 0; ks < KS1; ++ks) {
      v2f a = *(const v2f*)(Arow + ks * 4);
      v2f b = w1v[((size_t)ks * 32 + nc) * 32 + lane];
      h = wmma_f32(a, b, h);
    }
    const float bias = cb1[nc * 16 + col];
    #pragma unroll
    for (int r = 0; r < 8; ++r)
      hb[(Mloc0 + r) * 20 + col] = fmaxf(h[r] + bias, 0.f);
    #pragma unroll
    for (int k2 = 0; k2 < 4; ++k2) {
      v2f a2 = *(const v2f*)(hb + col * 20 + k2 * 4 + kadd);
      v2f b2 = w2v[(size_t)(nc * 4 + k2) * 32 + lane];
      cacc = wmma_f32(a2, b2, cacc);
    }
  }

  if (col < NCAT) {
    #pragma unroll
    for (int r = 0; r < 8; ++r)
      lg[Mloc0 + r][col] = cacc[r] + cb2[col];
  }
  __syncthreads();

  if (lane < 16) {
    float* o = out + OUT_CAT + (size_t)(g0 + lane) * NCAT;
    if (!vflag[lane]) {
      const float nanv = __int_as_float(0x7FC00000);
      for (int c = 0; c < NCAT; ++c) o[c] = nanv;
    } else {
      float mx = -3.4e38f;
      for (int c = 0; c < NCAT; ++c) mx = fmaxf(mx, lg[lane][c]);
      float e[NCAT], s = 0.f;
      for (int c = 0; c < NCAT; ++c) { e[c] = __expf(lg[lane][c] - mx); s += e[c]; }
      float inv = 1.f / s;
      for (int c = 0; c < NCAT; ++c) o[c] = e[c] * inv;
    }
  }
}

extern "C" void kernel_launch(void* const* d_in, const int* in_sizes, int n_in,
                              void* d_out, int out_size, void* d_ws, size_t ws_size,
                              hipStream_t stream) {
  const float* lhs    = (const float*)d_in[0];
  const int*   mask   = (const int*)  d_in[1];
  const float* cat_W1 = (const float*)d_in[2];
  const float* cat_b1 = (const float*)d_in[3];
  const float* cat_W2 = (const float*)d_in[4];
  const float* cat_b2 = (const float*)d_in[5];
  const float* ner_W1 = (const float*)d_in[6];
  const float* ner_b1 = (const float*)d_in[7];
  const float* ner_W2 = (const float*)d_in[8];
  const float* ner_b2 = (const float*)d_in[9];
  float* out = (float*)d_out;
  char*  wsb = (char*)d_ws;

  _Float16* nw1h = (_Float16*)(wsb + WSB_NW1H);
  _Float16* nw1l = (_Float16*)(wsb + WSB_NW1L);
  float*    cw1p = (float*)   (wsb + WSB_CW1);
  float*    nw2p = (float*)   (wsb + WSB_NW2);
  float*    cw2p = (float*)   (wsb + WSB_CW2);

  const int pf16_threads = KC1 * NC1 * 32;  // 24576
  const int pf32_threads = KS1 * NC1 * 32;  // 196608
  const int pw2_threads  = KS2 * 32;        // 4096
  pack_w1_f16_kernel<<<(pf16_threads + 255) / 256, 256, 0, stream>>>(ner_W1, nw1h, nw1l);
  pack_w1_f32_kernel<<<(pf32_threads + 255) / 256, 256, 0, stream>>>(cat_W1, cw1p);
  pack_w2_kernel<<<(pw2_threads + 255) / 256, 256, 0, stream>>>(ner_W2, NNER, nw2p);
  pack_w2_kernel<<<(pw2_threads + 255) / 256, 256, 0, stream>>>(cat_W2, NCAT, cw2p);

  ner_kernel<<<BSN, 256, 0, stream>>>(lhs, mask, ner_b1, ner_b2, nw1h, nw1l, nw2p, out);
  cat_kernel<<<BSN / 16, 32, 0, stream>>>(lhs, mask, cat_b1, cat_b2, cw1p, cw2p, out);
}
